// Transformer_50319836840018
// MI455X (gfx1250) — compile-verified, matrix operand order checked by reading, not run on hardware
//
#include <hip/hip_runtime.h>
#include <math.h>

// ---------------------------------------------------------------------------
// MI455X (gfx1250) transformer forward.
// Every GEMM (qkv / proj / fc1 / fc2 / QK^T / PV) runs on
// v_wmma_f32_16x16x32_bf16 (wave32, f32 accumulate). bf16 is produced once per
// tensor and every WMMA operand is fed by contiguous global_load_b128 fragment
// loads: the qkv GEMM epilogue scatters straight into attention-native
// layouts, Q/K [b,h][n][64] and V-transposed [b,h][64][n], so the PV B-matrix
// is k-contiguous too. fp32 only for residual stream / LN / softmax / GELU.
// ---------------------------------------------------------------------------

typedef __bf16 bf16_t;
typedef bf16_t v16bf __attribute__((ext_vector_type(16)));
typedef float  v8f   __attribute__((ext_vector_type(8)));
typedef unsigned short u16;

#define T_TOK      8192      // B*N tokens
#define CDIM       512
#define NSEQ       2048
#define BB         4
#define HH         8
#define DH         64
#define QH_ELEMS   ((size_t)BB * HH * NSEQ * DH)   // one of Q/K/Vt: 4 Mi elems

union Frag16 {
  v16bf v;
  u16   u[16];
  uint4 q[2];
};

__device__ __forceinline__ u16 f2bf(float f) {
  unsigned int u = __float_as_uint(f);
  u += 0x7FFFu + ((u >> 16) & 1u);          // round-to-nearest-even
  return (u16)(u >> 16);
}

__device__ __forceinline__ v8f v8f_zero() {
  v8f z;
#pragma unroll
  for (int e = 0; e < 8; ++e) z[e] = 0.0f;
  return z;
}

// bf16 16x32 (MxK) fragment: lane's row base `p`; lanes 0-15 hold K {0..7,
// 16..23}, lanes 16-31 hold K {8..15, 24..31} -> two contiguous b128 loads.
__device__ __forceinline__ void load_frag_bf16(Frag16& f, const u16* p, int hi) {
  const u16* q = p + hi * 8;
  f.q[0] = *(const uint4*)(q);
  f.q[1] = *(const uint4*)(q + 16);
}

// ---------------------------------------------------------------------------
// fp32 -> bf16 bulk convert (weights; once per launch). 8 elems/thread.
// ---------------------------------------------------------------------------
__global__ __launch_bounds__(256) void conv_f32_bf16(
    const float* __restrict__ src, u16* __restrict__ dst, int n)
{
  const int i = (blockIdx.x * 256 + threadIdx.x) * 8;
  if (i >= n) return;
  float4 a = *(const float4*)(src + i);
  float4 b = *(const float4*)(src + i + 4);
  uint4 o;
  o.x = (unsigned)f2bf(a.x) | ((unsigned)f2bf(a.y) << 16);
  o.y = (unsigned)f2bf(a.z) | ((unsigned)f2bf(a.w) << 16);
  o.z = (unsigned)f2bf(b.x) | ((unsigned)f2bf(b.y) << 16);
  o.w = (unsigned)f2bf(b.z) | ((unsigned)f2bf(b.w) << 16);
  *(uint4*)(dst + i) = o;
}

// ---------------------------------------------------------------------------
// GEMM: Y[T x Co] = act(Xbf[T x Ci] @ Wbf[Co x Ci]^T + bias) (+ res)
// One wave -> 32x64 tile: 2 A-frags x 4 B-frags -> 8 WMMAs / 32-K step,
// B fragments reused across both M subtiles. Inner loop: b128 loads + WMMA.
// act:      1 -> exact-erf GELU.
// out_mode: 0 -> fp32 out (+ optional fp32 residual)
//           1 -> bf16 out, row-major [T][Co]
//           2 -> qkv scatter: Q[b,h][n][64] | K[b,h][n][64] | Vt[b,h][64][n]
// ---------------------------------------------------------------------------
__global__ __launch_bounds__(256) void gemm_bf16_wmma(
    const u16* __restrict__ X, const u16* __restrict__ W,
    const float* __restrict__ bias, const float* __restrict__ res,
    void* __restrict__ Yv, int T, int Ci, int Co, int act, int out_mode)
{
  const int lane = threadIdx.x & 31;
  const int wid  = threadIdx.x >> 5;
  const int hi   = lane >> 4;
  const int lrow = lane & 15;
  const int n64  = Co >> 6;
  const int w    = blockIdx.x * 8 + wid;
  const int nt   = w % n64;
  const int mt   = w / n64;
  const int m0   = mt * 32;
  const int c0   = nt * 64;

  const u16* xrow0 = X + (size_t)(m0 + lrow) * Ci;
  const u16* xrow1 = X + (size_t)(m0 + 16 + lrow) * Ci;
  const u16* wrow[4];
#pragma unroll
  for (int j = 0; j < 4; ++j)
    wrow[j] = W + (size_t)(c0 + j * 16 + lrow) * Ci;

  v8f acc[2][4];
#pragma unroll
  for (int mi = 0; mi < 2; ++mi)
#pragma unroll
    for (int j = 0; j < 4; ++j) acc[mi][j] = v8f_zero();

  for (int k0 = 0; k0 < Ci; k0 += 32) {
    Frag16 a0, a1;
    load_frag_bf16(a0, xrow0 + k0, hi);
    load_frag_bf16(a1, xrow1 + k0, hi);
#pragma unroll
    for (int j = 0; j < 4; ++j) {
      Frag16 b;
      load_frag_bf16(b, wrow[j] + k0, hi);
      acc[0][j] = __builtin_amdgcn_wmma_f32_16x16x32_bf16(
          false, a0.v, false, b.v, (short)0, acc[0][j], false, false);
      acc[1][j] = __builtin_amdgcn_wmma_f32_16x16x32_bf16(
          false, a1.v, false, b.v, (short)0, acc[1][j], false, false);
    }
  }

  // Epilogue. C/D layout: lanes 0-15 rows 0-7, lanes 16-31 rows 8-15.
#pragma unroll
  for (int mi = 0; mi < 2; ++mi) {
#pragma unroll
    for (int j = 0; j < 4; ++j) {
      const int cbase = c0 + j * 16;
      const int co    = cbase + lrow;
      const float bz  = bias ? bias[co] : 0.0f;
      // qkv scatter coordinates (uniform per j-tile except d's lrow part)
      const int ss = cbase >> 9;            // 0=Q 1=K 2=V
      const int hh = (cbase >> 6) & 7;      // head
      const int d  = (cbase & 63) + lrow;   // 0..63
#pragma unroll
      for (int e = 0; e < 8; ++e) {
        const int r = m0 + mi * 16 + e + hi * 8;   // token index
        float v = acc[mi][j][e] + bz;
        if (act == 1)
          v = 0.5f * v * (1.0f + erff(v * 0.70710678118654752f));
        if (out_mode == 0) {
          const size_t idx = (size_t)r * Co + co;
          if (res) v += res[idx];
          ((float*)Yv)[idx] = v;
        } else if (out_mode == 1) {
          ((u16*)Yv)[(size_t)r * Co + co] = f2bf(v);
        } else {
          const int bh = ((r >> 11) << 3) + hh;    // b*8 + h
          const int n  = r & (NSEQ - 1);
          size_t idx;
          if (ss == 2)   // V transposed: [bh][d][n]
            idx = 2 * QH_ELEMS + ((size_t)bh * DH + d) * NSEQ + n;
          else           // Q / K: [bh][n][d]
            idx = (size_t)ss * QH_ELEMS + ((size_t)bh * NSEQ + n) * DH + d;
          ((u16*)Yv)[idx] = f2bf(v);
        }
      }
    }
  }
}

// ---------------------------------------------------------------------------
// Flash-style attention: one wave per (b, h, 16-query-row tile).
// All operands in attention-native bf16 layouts -> every fragment is two
// global_load_b128s. S = Q K^T (4 WMMA), online fp32 softmax (shfl_xor row
// reductions), P C-layout -> A-layout via 1KB wave-private LDS, O += P V
// (4 WMMA). Emits bf16 O [T][C] for the proj GEMM.
// ---------------------------------------------------------------------------
__global__ __launch_bounds__(128) void attn_wmma(
    const u16* __restrict__ qb, const u16* __restrict__ kb_,
    const u16* __restrict__ vt, u16* __restrict__ o)
{
  __shared__ __align__(16) u16 lds_p[4][16 * 32];

  const int lane = threadIdx.x & 31;
  const int wib  = threadIdx.x >> 5;
  const int hi   = lane >> 4;
  const int lrow = lane & 15;
  const int w    = blockIdx.x * 4 + wib;
  const int qt   = w & 127;          // 2048/16 query tiles
  const int bh   = w >> 7;           // b*8 + h
  const int h    = bh & 7;
  const int b    = bh >> 3;

  const u16* qrow = qb + ((size_t)bh * NSEQ + qt * 16 + lrow) * DH;
  Frag16 qf0, qf1;
  load_frag_bf16(qf0, qrow, hi);          // d = 0..31
  load_frag_bf16(qf1, qrow + 32, hi);     // d = 32..63

  v8f oa[4];
#pragma unroll
  for (int j = 0; j < 4; ++j) oa[j] = v8f_zero();
  float m[8], l[8];
#pragma unroll
  for (int e = 0; e < 8; ++e) { m[e] = -INFINITY; l[e] = 0.0f; }

  for (int kt = 0; kt < NSEQ / 32; ++kt) {
    const int kb = kt * 32;

    // --- scores S[16 x 32] as two 16x16 WMMA tiles (K-dim = Dh = 64) ---
    v8f s0 = v8f_zero(), s1 = v8f_zero();
    {
      const u16* krow = kb_ + ((size_t)bh * NSEQ + kb + lrow) * DH;
      Frag16 k00, k01, k10, k11;
      load_frag_bf16(k00, krow, hi);
      load_frag_bf16(k01, krow + 32, hi);
      const u16* krow2 = krow + (size_t)16 * DH;
      load_frag_bf16(k10, krow2, hi);
      load_frag_bf16(k11, krow2 + 32, hi);
      s0 = __builtin_amdgcn_wmma_f32_16x16x32_bf16(false, qf0.v, false, k00.v, (short)0, s0, false, false);
      s0 = __builtin_amdgcn_wmma_f32_16x16x32_bf16(false, qf1.v, false, k01.v, (short)0, s0, false, false);
      s1 = __builtin_amdgcn_wmma_f32_16x16x32_bf16(false, qf0.v, false, k10.v, (short)0, s1, false, false);
      s1 = __builtin_amdgcn_wmma_f32_16x16x32_bf16(false, qf1.v, false, k11.v, (short)0, s1, false, false);
    }

    // --- online softmax (rows in VGPR index, 16 cols across a lane-half) ---
#pragma unroll
    for (int e = 0; e < 8; ++e) {
      float sv0 = s0[e] * 0.125f;   // Dh^-0.5
      float sv1 = s1[e] * 0.125f;
      float mx = fmaxf(sv0, sv1);
      mx = fmaxf(mx, __shfl_xor(mx, 1, 32));
      mx = fmaxf(mx, __shfl_xor(mx, 2, 32));
      mx = fmaxf(mx, __shfl_xor(mx, 4, 32));
      mx = fmaxf(mx, __shfl_xor(mx, 8, 32));
      const float mnew  = fmaxf(m[e], mx);
      const float alpha = __expf(m[e] - mnew);
      const float p0 = __expf(sv0 - mnew);
      const float p1 = __expf(sv1 - mnew);
      float rs = p0 + p1;
      rs += __shfl_xor(rs, 1, 32);
      rs += __shfl_xor(rs, 2, 32);
      rs += __shfl_xor(rs, 4, 32);
      rs += __shfl_xor(rs, 8, 32);
      l[e] = l[e] * alpha + rs;
      m[e] = mnew;
#pragma unroll
      for (int j = 0; j < 4; ++j) oa[j][e] *= alpha;
      const int prow = e + hi * 8;
      lds_p[wib][prow * 32 + lrow]      = f2bf(p0);
      lds_p[wib][prow * 32 + 16 + lrow] = f2bf(p1);
    }
    __syncthreads();

    // --- reload P as a 16x32 bf16 A fragment ---
    Frag16 pf;
    {
      const u16* pp = &lds_p[wib][lrow * 32 + hi * 8];
      pf.q[0] = *(const uint4*)(pp);
      pf.q[1] = *(const uint4*)(pp + 16);
    }
    __syncthreads();

    // --- O += P V : 4 WMMAs across Dh; Vt is k-contiguous -> b128 loads ---
    const u16* vrow = vt + ((size_t)bh * DH + lrow) * NSEQ + kb;
#pragma unroll
    for (int j = 0; j < 4; ++j) {
      Frag16 vf;
      load_frag_bf16(vf, vrow + (size_t)(j * 16) * NSEQ, hi);
      oa[j] = __builtin_amdgcn_wmma_f32_16x16x32_bf16(
          false, pf.v, false, vf.v, (short)0, oa[j], false, false);
    }
  }

  // --- normalize and write O as bf16 [T x C] (heads concatenated) ---
#pragma unroll
  for (int j = 0; j < 4; ++j) {
#pragma unroll
    for (int e = 0; e < 8; ++e) {
      const int t = b * NSEQ + qt * 16 + e + hi * 8;
      const int c = h * DH + j * 16 + lrow;
      o[(size_t)t * CDIM + c] = f2bf(oa[j][e] / l[e]);
    }
  }
}

// ---------------------------------------------------------------------------
// LayerNorm over C=512: one block per token; fp32 in, bf16 out (GEMM input).
// ---------------------------------------------------------------------------
__global__ __launch_bounds__(256) void ln_kernel(
    const float* __restrict__ x, const float* __restrict__ w,
    const float* __restrict__ b, u16* __restrict__ y)
{
  const int t   = blockIdx.x;
  const int tid = threadIdx.x;
  const float* xr = x + (size_t)t * CDIM;
  float2 v = *(const float2*)(xr + tid * 2);
  float s  = v.x + v.y;
  float ss = v.x * v.x + v.y * v.y;
#pragma unroll
  for (int off = 16; off >= 1; off >>= 1) {
    s  += __shfl_down(s, off, 32);
    ss += __shfl_down(ss, off, 32);
  }
  __shared__ float sh[18];
  const int wid = tid >> 5;
  if ((tid & 31) == 0) { sh[wid] = s; sh[wid + 8] = ss; }
  __syncthreads();
  if (tid == 0) {
    float S = 0.0f, SS = 0.0f;
#pragma unroll
    for (int i = 0; i < 8; ++i) { S += sh[i]; SS += sh[i + 8]; }
    const float mean = S * (1.0f / CDIM);
    const float var  = SS * (1.0f / CDIM) - mean * mean;
    sh[16] = mean;
    sh[17] = rsqrtf(var + 1e-5f);
  }
  __syncthreads();
  const float mean = sh[16], inv = sh[17];
  const int i0 = tid * 2;
  float2 wv = *(const float2*)(w + i0);
  float2 bv = *(const float2*)(b + i0);
  const u16 h0 = f2bf((v.x - mean) * inv * wv.x + bv.x);
  const u16 h1 = f2bf((v.y - mean) * inv * wv.y + bv.y);
  *(unsigned int*)(y + (size_t)t * CDIM + i0) =
      (unsigned)h0 | ((unsigned)h1 << 16);
}

// ---------------------------------------------------------------------------
// Orchestration (graph-capture safe). Workspace (~67 MB):
//   bf16 weights (24 MB, converted once per launch) | h/o/h2 bf16 (8 MB) |
//   Q|K|Vt (24 MB) / MLP-hidden (32 MB) shared region.
// Residual stream x lives fp32 in d_out.
// ---------------------------------------------------------------------------
extern "C" void kernel_launch(void* const* d_in, const int* in_sizes, int n_in,
                              void* d_out, int out_size, void* d_ws, size_t ws_size,
                              hipStream_t stream) {
  const float* x_in   = (const float*)d_in[0];
  const float* ln1_w  = (const float*)d_in[1];
  const float* ln1_b  = (const float*)d_in[2];
  const float* qkv_w  = (const float*)d_in[3];
  const float* proj_w = (const float*)d_in[4];
  const float* proj_b = (const float*)d_in[5];
  const float* ln2_w  = (const float*)d_in[6];
  const float* ln2_b  = (const float*)d_in[7];
  const float* fc1_w  = (const float*)d_in[8];
  const float* fc1_b  = (const float*)d_in[9];
  const float* fc2_w  = (const float*)d_in[10];
  const float* fc2_b  = (const float*)d_in[11];

  char* ws = (char*)d_ws;
  size_t off = 0;
  u16* wqkv = (u16*)(ws + off); off += (size_t)4 * 1536 * CDIM * 2;
  u16* wprj = (u16*)(ws + off); off += (size_t)4 * CDIM * CDIM * 2;
  u16* wfc1 = (u16*)(ws + off); off += (size_t)4 * 2048 * CDIM * 2;
  u16* wfc2 = (u16*)(ws + off); off += (size_t)4 * CDIM * 2048 * 2;
  u16* hbuf = (u16*)(ws + off); off += (size_t)T_TOK * CDIM * 2;
  u16* big  = (u16*)(ws + off); off += (size_t)T_TOK * 2048 * 2;
  // Q|K|Vt views inside `big` (3 * 8 MB <= 32 MB):
  u16* qbuf = big;
  u16* kbuf = big + QH_ELEMS;
  u16* vtbf = big + 2 * QH_ELEMS;

  float* x = (float*)d_out;
  hipMemcpyAsync(x, x_in, (size_t)T_TOK * CDIM * sizeof(float),
                 hipMemcpyDeviceToDevice, stream);

  // One-time bf16 weight conversion (all element counts divide 2048 exactly).
  conv_f32_bf16<<<(4 * 1536 * CDIM) / 2048, 256, 0, stream>>>(qkv_w,  wqkv, 4 * 1536 * CDIM);
  conv_f32_bf16<<<(4 * CDIM * CDIM) / 2048, 256, 0, stream>>>(proj_w, wprj, 4 * CDIM * CDIM);
  conv_f32_bf16<<<(4 * 2048 * CDIM) / 2048, 256, 0, stream>>>(fc1_w,  wfc1, 4 * 2048 * CDIM);
  conv_f32_bf16<<<(4 * CDIM * 2048) / 2048, 256, 0, stream>>>(fc2_w,  wfc2, 4 * CDIM * 2048);

  for (int i = 0; i < 4; ++i) {
    // --- attention block ---
    ln_kernel<<<T_TOK, 256, 0, stream>>>(x, ln1_w + i * CDIM, ln1_b + i * CDIM, hbuf);
    gemm_bf16_wmma<<<(T_TOK / 32) * (1536 / 64) / 8, 256, 0, stream>>>(
        hbuf, wqkv + (size_t)i * 1536 * CDIM, nullptr, nullptr, big,
        T_TOK, CDIM, 1536, 0, 2 /* qkv scatter */);
    attn_wmma<<<(BB * HH * (NSEQ / 16)) / 4, 128, 0, stream>>>(qbuf, kbuf, vtbf, hbuf);
    gemm_bf16_wmma<<<(T_TOK / 32) * (CDIM / 64) / 8, 256, 0, stream>>>(
        hbuf, wprj + (size_t)i * CDIM * CDIM, proj_b + i * CDIM, x, x,
        T_TOK, CDIM, CDIM, 0, 0 /* fp32 out + residual */);

    // --- MLP block ---
    ln_kernel<<<T_TOK, 256, 0, stream>>>(x, ln2_w + i * CDIM, ln2_b + i * CDIM, hbuf);
    gemm_bf16_wmma<<<(T_TOK / 32) * (2048 / 64) / 8, 256, 0, stream>>>(
        hbuf, wfc1 + (size_t)i * 2048 * CDIM, fc1_b + i * 2048, nullptr, big,
        T_TOK, CDIM, 2048, 1 /* GELU */, 1 /* bf16 out */);
    gemm_bf16_wmma<<<(T_TOK / 32) * (CDIM / 64) / 8, 256, 0, stream>>>(
        big, wfc2 + (size_t)i * CDIM * 2048, fc2_b + i * CDIM, x, x,
        T_TOK, 2048, CDIM, 0, 0 /* fp32 out + residual */);
  }
}
